// RGAT_50525995270158
// MI455X (gfx1250) — matrix operand order repaired and checked by reading, not compiled, not run on hardware
//
#include <hip/hip_runtime.h>

// ---------------------------------------------------------------------------
// RGAT (2-layer, heads=1) for MI455X / gfx1250, wave32.
// Dense per-node per-relation transform runs on v_wmma_f32_16x16x32_f16 with
// vectorized (b128) LDS fragment loads and a staged, coalesced b128 epilogue.
// Edge phase is bandwidth-bound; xw kept in f16 (51 MB -> resident in L2).
// ---------------------------------------------------------------------------

typedef __attribute__((ext_vector_type(16))) _Float16 v16h;
typedef __attribute__((ext_vector_type(8)))  _Float16 v8h;
typedef __attribute__((ext_vector_type(4)))  _Float16 v4h;
typedef __attribute__((ext_vector_type(8)))  float    v8f;

#define RREL 8
#define DIM  64
#define NEG_SLOPE 0.2f
#define EPS_SM 1e-16f

// ---------------- WMMA transform: xw[n,r,:] = x[n,:] @ W[r] ----------------
// grid = (ceil(N/16), R), block = 128 (4 waves). Wave w computes out cols
// [16w,16w+16) of a 16-node tile. K=64 split into two K=32 WMMA steps.
//
// Fragment layouts used here (A/B share one k order), exploited for contiguity:
//  A (16x32 f16): lane(m,half): e=0..7  -> k = 16*half + e       (16B run)
//                               e=8..15 -> k = 16*half + e       (16B run)
//  B (32x16 f16): lane(n,half): e=0..15 -> k = 16*half + e       (two 16B runs)
//  D (16x16 f32): vgpr i -> row m = i + 8*half, col n = lane%16
__global__ __launch_bounds__(128)
void rgat_transform(const float* __restrict__ xin, const float* __restrict__ W,
                    _Float16* __restrict__ xw, int N) {
  __shared__ _Float16 sX [16][DIM] __attribute__((aligned(16)));  // 2 KB, [m][k]
  __shared__ _Float16 sWt[DIM][DIM] __attribute__((aligned(16))); // 8 KB, [col][k]
  __shared__ _Float16 sD [16][DIM] __attribute__((aligned(16)));  // 2 KB, [m][col]
  const int r    = blockIdx.y;
  const int row0 = blockIdx.x * 16;
  const int tid  = threadIdx.x;

  // Stage W[r] transposed: W is [k][col] in memory -> sWt[col][k].
  const float4* Wr4 = (const float4*)(W + (size_t)r * DIM * DIM);
  for (int i = tid; i < (DIM * DIM) / 4; i += 128) {
    float4 v = Wr4[i];
    int kk = (i * 4) >> 6;   // K row
    int c  = (i * 4) & 63;   // col base
    sWt[c + 0][kk] = (_Float16)v.x;
    sWt[c + 1][kk] = (_Float16)v.y;
    sWt[c + 2][kk] = (_Float16)v.z;
    sWt[c + 3][kk] = (_Float16)v.w;
  }
  // Stage x tile row-major: sX[m][k].
  const float4* x4 = (const float4*)xin;
  for (int i = tid; i < (16 * DIM) / 4; i += 128) {
    int m = (i * 4) >> 6, c = (i * 4) & 63, n = row0 + m;
    float4 v = (n < N) ? x4[(size_t)n * (DIM / 4) + (c >> 2)]
                       : float4{0.f, 0.f, 0.f, 0.f};
    v4h h; h[0] = (_Float16)v.x; h[1] = (_Float16)v.y;
           h[2] = (_Float16)v.z; h[3] = (_Float16)v.w;
    *(v4h*)&sX[m][c] = h;
  }
  __syncthreads();

  const int wave = tid >> 5, lane = tid & 31;
  const int half = lane >> 4, lm = lane & 15;
  const int col  = wave * 16 + lm;

  v8f c = {};
#pragma unroll
  for (int kh = 0; kh < 2; ++kh) {
    const int ko = kh * 32;
    v8h a0 = *(const v8h*)&sX[lm][ko + 16 * half];       // k = ko+16h .. +7
    v8h a1 = *(const v8h*)&sX[lm][ko + 16 * half + 8];   // k = ko+16h+8 .. +7
    v8h b0 = *(const v8h*)&sWt[col][ko + 16 * half];     // k = ko+16h .. +7
    v8h b1 = *(const v8h*)&sWt[col][ko + 16 * half + 8]; // k = ko+16h+8 .. +7
    v16h a = __builtin_shufflevector(a0, a1, 0, 1, 2, 3, 4, 5, 6, 7,
                                     8, 9, 10, 11, 12, 13, 14, 15);
    v16h b = __builtin_shufflevector(b0, b1, 0, 1, 2, 3, 4, 5, 6, 7,
                                     8, 9, 10, 11, 12, 13, 14, 15);
    c = __builtin_amdgcn_wmma_f32_16x16x32_f16(false, a, false, b,
                                               (short)0, c, false, false);
  }

  // Stage D tile in LDS (f16), then write out coalesced.
#pragma unroll
  for (int i = 0; i < 8; ++i)
    sD[i + half * 8][col] = (_Float16)c[i];
  __syncthreads();

  // 128 threads: 8 threads per node row, one global_store_b128 each.
  {
    int m = tid >> 3, chunk = tid & 7;
    int n = row0 + m;
    if (n < N) {
      v8h d = *(const v8h*)&sD[m][chunk * 8];
      *(v8h*)(xw + ((size_t)n * RREL + r) * DIM + chunk * 8) = d;
    }
  }
}

// --------------- per-(node,relation) attention scalars qn, kn ---------------
__global__ void rgat_node_dots(const _Float16* __restrict__ xw,
                               const float* __restrict__ q,
                               const float* __restrict__ k,
                               float* __restrict__ qn, float* __restrict__ kn,
                               int NR) {
  int idx = blockIdx.x * blockDim.x + threadIdx.x;
  if (idx >= NR) return;
  const v8h* row = (const v8h*)(xw + (size_t)idx * DIM);
  float sq = 0.f, sk = 0.f;
#pragma unroll
  for (int ch = 0; ch < 8; ++ch) {
    v8h v = row[ch];
#pragma unroll
    for (int j = 0; j < 8; ++j) {
      float f = (float)v[j];
      sq += f * q[ch * 8 + j];
      sk += f * k[ch * 8 + j];
    }
  }
  qn[idx] = sq;
  kn[idx] = sk;
}

// ------------------------- per-layer state init ----------------------------
__global__ void rgat_init(float* __restrict__ acc, const float* __restrict__ bias,
                          unsigned* __restrict__ mkey, float* __restrict__ ssum,
                          int N) {
  int idx = blockIdx.x * blockDim.x + threadIdx.x;
  if (idx < N * DIM) acc[idx] = bias[idx & 63];
  if (idx < N) { mkey[idx] = 0u; ssum[idx] = 0.0f; }
}

// monotone float<->uint map so atomicMax(u32) == float max
__device__ __forceinline__ unsigned fkey(float f) {
  unsigned u = __float_as_uint(f);
  return (u & 0x80000000u) ? ~u : (u | 0x80000000u);
}
__device__ __forceinline__ float funkey(unsigned u) {
  return __uint_as_float((u & 0x80000000u) ? (u ^ 0x80000000u) : ~u);
}

// ---------------- edge logits + segment max over dst -----------------------
__global__ void rgat_logits(const int* __restrict__ src, const int* __restrict__ dst,
                            const int* __restrict__ et,
                            const float* __restrict__ qn, const float* __restrict__ kn,
                            float* __restrict__ logit, unsigned* __restrict__ mkey,
                            int E) {
  int e = blockIdx.x * blockDim.x + threadIdx.x;
  if (e >= E) return;
  int r = et[e], s = src[e], d = dst[e];
  float l = qn[d * RREL + r] + kn[s * RREL + r];
  l = (l > 0.f) ? l : NEG_SLOPE * l;     // leaky_relu
  logit[e] = l;
  atomicMax(&mkey[d], fkey(l));
}

// ---------------- exp(logit - m[dst]) + segment sum ------------------------
__global__ void rgat_expsum(const int* __restrict__ dst,
                            const float* __restrict__ logit,
                            const unsigned* __restrict__ mkey,
                            float* __restrict__ ebuf, float* __restrict__ ssum,
                            int E) {
  int e = blockIdx.x * blockDim.x + threadIdx.x;
  if (e >= E) return;
  int d = dst[e];
  unsigned mk = mkey[d];
  float m = (mk == 0u) ? 0.0f : funkey(mk);
  if (!isfinite(m)) m = 0.0f;            // reference's where(isfinite(m), m, 0)
  float ev = __expf(logit[e] - m);
  ebuf[e] = ev;
  atomicAdd(&ssum[d], ev);
}

// --------- alpha = e/(s+eps); acc[dst,:] += alpha * xw[src,etype,:] --------
// 64 threads per edge (consecutive o -> coalesced); xw gathers hit L2.
__global__ void rgat_scatter(const int* __restrict__ src, const int* __restrict__ dst,
                             const int* __restrict__ et,
                             const _Float16* __restrict__ xw,
                             const float* __restrict__ ebuf,
                             const float* __restrict__ ssum,
                             float* __restrict__ acc, float* __restrict__ alpha_out,
                             long long EW) {
  long long idx = (long long)blockIdx.x * blockDim.x + threadIdx.x;
  if (idx >= EW) return;
  int e = (int)(idx >> 6), o = (int)(idx & 63);
  int d = dst[e];
  float alpha = ebuf[e] / (ssum[d] + EPS_SM);
  if (o == 0 && alpha_out) alpha_out[e] = alpha;
  int s = src[e], r = et[e];
  float v = (float)xw[((size_t)s * RREL + r) * DIM + o];
  atomicAdd(&acc[(size_t)d * DIM + o], alpha * v);
}

__global__ void rgat_relu(const float* __restrict__ acc, float* __restrict__ out,
                          int n) {
  int i = blockIdx.x * blockDim.x + threadIdx.x;
  if (i < n) out[i] = fmaxf(acc[i], 0.0f);
}

__global__ void rgat_copy_i32(const int* __restrict__ in, int* __restrict__ out,
                              int n) {
  int i = blockIdx.x * blockDim.x + threadIdx.x;
  if (i < n) out[i] = in[i];
}

// ---------------------------------------------------------------------------
static void run_layer(const float* xin, const float* W, const float* q,
                      const float* k, const float* bias,
                      const int* src, const int* dst, const int* et,
                      int N, int E,
                      _Float16* xw, float* qn, float* kn, float* logit,
                      float* ebuf, unsigned* mkey, float* ssum, float* acc,
                      float* hout, float* alpha_out, hipStream_t stream) {
  dim3 gT((N + 15) / 16, RREL);
  rgat_transform<<<gT, 128, 0, stream>>>(xin, W, xw, N);
  int NR = N * RREL;
  rgat_node_dots<<<(NR + 255) / 256, 256, 0, stream>>>(xw, q, k, qn, kn, NR);
  int NI = N * DIM;
  rgat_init<<<(NI + 255) / 256, 256, 0, stream>>>(acc, bias, mkey, ssum, N);
  rgat_logits<<<(E + 255) / 256, 256, 0, stream>>>(src, dst, et, qn, kn, logit,
                                                   mkey, E);
  rgat_expsum<<<(E + 255) / 256, 256, 0, stream>>>(dst, logit, mkey, ebuf, ssum, E);
  long long EW = (long long)E * DIM;
  rgat_scatter<<<(unsigned)((EW + 255) / 256), 256, 0, stream>>>(
      src, dst, et, xw, ebuf, ssum, acc, alpha_out, EW);
  rgat_relu<<<(NI + 255) / 256, 256, 0, stream>>>(acc, hout, NI);
}

extern "C" void kernel_launch(void* const* d_in, const int* in_sizes, int n_in,
                              void* d_out, int out_size, void* d_ws, size_t ws_size,
                              hipStream_t stream) {
  (void)n_in; (void)out_size; (void)ws_size;
  const float* x   = (const float*)d_in[0];
  const int*   ei  = (const int*)d_in[1];
  const int*   et  = (const int*)d_in[2];
  const float* W1  = (const float*)d_in[3];
  const float* q1  = (const float*)d_in[4];
  const float* k1  = (const float*)d_in[5];
  const float* b1  = (const float*)d_in[6];
  const float* W2  = (const float*)d_in[7];
  const float* q2  = (const float*)d_in[8];
  const float* k2  = (const float*)d_in[9];
  const float* b2  = (const float*)d_in[10];

  const int N = in_sizes[0] / DIM;      // 50000
  const int E = in_sizes[2];            // 800000
  const int* src = ei;                  // row 0
  const int* dst = ei + E;              // row 1

  // ---- carve workspace (256B aligned) ----
  char* wp = (char*)d_ws;
  auto carve = [&](size_t bytes) -> void* {
    void* p = (void*)wp;
    wp += (bytes + 255) & ~(size_t)255;
    return p;
  };
  _Float16* xw    = (_Float16*)carve((size_t)N * RREL * DIM * sizeof(_Float16));
  float*    qn    = (float*)   carve((size_t)N * RREL * sizeof(float));
  float*    kn    = (float*)   carve((size_t)N * RREL * sizeof(float));
  float*    logit = (float*)   carve((size_t)E * sizeof(float));
  float*    ebuf  = (float*)   carve((size_t)E * sizeof(float));
  unsigned* mkey  = (unsigned*)carve((size_t)N * sizeof(unsigned));
  float*    ssum  = (float*)   carve((size_t)N * sizeof(float));
  float*    acc   = (float*)   carve((size_t)N * DIM * sizeof(float));
  float*    h1    = (float*)   carve((size_t)N * DIM * sizeof(float));

  // ---- output layout: h [N*64 f32] | edge_index [2E i32 bits] | alpha [E f32]
  float* out_h     = (float*)d_out;
  int*   out_ei    = (int*)d_out + (size_t)N * DIM;
  float* out_alpha = (float*)d_out + (size_t)N * DIM + (size_t)2 * E;

  // layer 1 (alpha discarded)
  run_layer(x, W1, q1, k1, b1, src, dst, et, N, E,
            xw, qn, kn, logit, ebuf, mkey, ssum, acc, h1, nullptr, stream);
  // layer 2 (h -> d_out, alpha -> d_out)
  run_layer(h1, W2, q2, k2, b2, src, dst, et, N, E,
            xw, qn, kn, logit, ebuf, mkey, ssum, acc, out_h, out_alpha, stream);

  int EI = 2 * E;
  rgat_copy_i32<<<(EI + 255) / 256, 256, 0, stream>>>(ei, out_ei, EI);
}
